// CMGCA_13683765805779
// MI455X (gfx1250) — matrix operand order repaired and checked by reading, not compiled, not run on hardware
//
#include <hip/hip_runtime.h>
#include <math.h>

// ---------------------------------------------------------------------------
// CMGCA fused cross-attention for MI455X (gfx1250, wave32, WMMA bf16)
//
// Algebraic collapse:
//   G_b   = A_b @ B_b^T                      (256x256, K=4096 spatial)
//   logitsA_h = s * Wq_h G Wk_h^T ; logitsB_h = s * Wq_h G^T Wk_h^T
//   GM    = diag(g) * (softmax(logits) @ Wv_h)   (per dir, 256x256)
//   P     = Wproj @ GM                       (256x256)
//   out   = base + P_b @ X_flat              (256x4096, fused residual)
//
// Round-2 fixes: spread k_gram over 64 blocks (4-way column split; L2 holds
// A+B so the re-reads are free at HBM level) and split pooling into a
// 128-block atomic partial-sum pass + tiny gate MLP, so no stage runs on
// only 16 WGPs.
// ---------------------------------------------------------------------------

typedef __bf16 bf16;
typedef __attribute__((ext_vector_type(16))) __bf16 bf16x16;
typedef __attribute__((ext_vector_type(8)))  __bf16 bf16x8;
typedef __attribute__((ext_vector_type(8)))  float  v8f;

#define BB 16
#define CC 256
#define NN 4096
#define NH 4
#define CH 64
#define PITCH 40           // bf16 elements per row of a 32-wide LDS tile

static __device__ __forceinline__ v8f wmma_bf16(bf16x16 a, bf16x16 b, v8f c) {
  return __builtin_amdgcn_wmma_f32_16x16x32_bf16(false, a, false, b, (short)0, c,
                                                 false, false);
}

static __device__ __forceinline__ v8f v8f_zero() {
  v8f z = {0.f, 0.f, 0.f, 0.f, 0.f, 0.f, 0.f, 0.f};
  return z;
}

// --- fragment loaders -------------------------------------------------------
// A-operand 16x32 layout (bf16): lane<16 -> row r0+lane, elems 0..7 = K k0..k0+7,
// elems 8..15 = K k0+16..23 ; lane>=16 -> same rows, K +8 / +24.
// B-operand 32x16 layout: lane<16 -> col n0+lane, elems 0..7 = K k0..k0+7,
// elems 8..15 = K k0+16..23 ; lane>=16 -> same cols, K +8 / +24.

static __device__ __forceinline__ bf16x16 ldsA40(const bf16* t, int r0, int lane) {
  const bf16* p = t + (r0 + (lane & 15)) * PITCH + (lane >> 4) * 8;
  bf16x8 lo = *(const bf16x8*)p;
  bf16x8 hi = *(const bf16x8*)(p + 16);
  bf16x16 r;
#pragma unroll
  for (int i = 0; i < 8; ++i) { r[i] = lo[i]; r[8 + i] = hi[i]; }
  return r;
}

static __device__ __forceinline__ bf16x16 ldsB40(const bf16* t, int n0, int lane) {
  const bf16* p = t + (n0 + (lane & 15)) * PITCH + (lane >> 4) * 8;
  bf16x8 lo = *(const bf16x8*)p;
  bf16x8 hi = *(const bf16x8*)(p + 16);
  bf16x16 r;
#pragma unroll
  for (int i = 0; i < 8; ++i) { r[i] = lo[i]; r[8 + i] = hi[i]; }
  return r;
}

static __device__ __forceinline__ bf16x16 ldsA(const bf16* t, int r0, int pitch,
                                               int k0, int lane) {
  const bf16* p = t + (r0 + (lane & 15)) * pitch + k0 + (lane >> 4) * 8;
  bf16x8 lo = *(const bf16x8*)p;
  bf16x8 hi = *(const bf16x8*)(p + 16);
  bf16x16 r;
#pragma unroll
  for (int i = 0; i < 8; ++i) { r[i] = lo[i]; r[8 + i] = hi[i]; }
  return r;
}

// A-operand straight from global fp32 row-major [rows][ld]
static __device__ __forceinline__ bf16x16 gloA(const float* base, int r0, int ld,
                                               int k0, int lane) {
  const float* p = base + (size_t)(r0 + (lane & 15)) * ld + k0 + (lane >> 4) * 8;
  bf16x16 r;
#pragma unroll
  for (int i = 0; i < 8; ++i) { r[i] = (bf16)p[i]; r[8 + i] = (bf16)p[16 + i]; }
  return r;
}

// A-operand from global bf16 row-major [rows][ld]
static __device__ __forceinline__ bf16x16 gbfA(const bf16* base, int r0, int ld,
                                               int k0, int lane) {
  const bf16* p = base + (size_t)(r0 + (lane & 15)) * ld + k0 + (lane >> 4) * 8;
  bf16x8 lo = *(const bf16x8*)p;
  bf16x8 hi = *(const bf16x8*)(p + 16);
  bf16x16 r;
#pragma unroll
  for (int i = 0; i < 8; ++i) { r[i] = lo[i]; r[8 + i] = hi[i]; }
  return r;
}

// B-operand where element(k,n) = src[n0+n][k0+k]  (rows of src, contiguous k)
static __device__ __forceinline__ bf16x16 gloBrows(const float* base, int n0, int ld,
                                                   int k0, int lane) {
  const float* p = base + (size_t)(n0 + (lane & 15)) * ld + k0 + (lane >> 4) * 8;
  bf16x16 r;
#pragma unroll
  for (int i = 0; i < 16; ++i) r[i] = (bf16)p[i < 8 ? i : i + 8];
  return r;
}

// B-operand where element(k,n) = src[k0+k][n0+n]  (columns, strided by ld)
static __device__ __forceinline__ bf16x16 gloBcols(const float* base, int k0, int ld,
                                                   int n0, int lane) {
  const float* p = base + (size_t)(k0 + (lane >> 4) * 8) * ld + n0 + (lane & 15);
  bf16x16 r;
#pragma unroll
  for (int i = 0; i < 16; ++i) r[i] = (bf16)p[(size_t)(i < 8 ? i : i + 8) * ld];
  return r;
}

// ---------------------------------------------------------------------------
// K0: partial spatial sums -> poolA/poolB[b][256]  (grid 16x8, atomics)
// ---------------------------------------------------------------------------
__global__ void __launch_bounds__(256, 1)
k_pool(const float* __restrict__ A, const float* __restrict__ B,
       float* __restrict__ poolA, float* __restrict__ poolB) {
  int b = blockIdx.x, s = blockIdx.y, t = threadIdx.x;
  const float4* pa = (const float4*)(A + ((size_t)b * CC + t) * NN + s * (NN / 8));
  const float4* pb = (const float4*)(B + ((size_t)b * CC + t) * NN + s * (NN / 8));
  float sa = 0.f, sb = 0.f;
  for (int i = 0; i < NN / 32; ++i) {
    float4 x = pa[i]; sa += x.x + x.y + x.z + x.w;
    float4 y = pb[i]; sb += y.x + y.y + y.z + y.w;
  }
  atomicAdd(&poolA[b * CC + t], sa);
  atomicAdd(&poolB[b * CC + t], sb);
}

// ---------------------------------------------------------------------------
// K1: gate MLP -> g[b][256]
// ---------------------------------------------------------------------------
__global__ void __launch_bounds__(256, 1)
k_gate(const float* __restrict__ poolA, const float* __restrict__ poolB,
       const float* __restrict__ wg1, const float* __restrict__ bg1,
       const float* __restrict__ wg2, const float* __restrict__ bg2,
       float* __restrict__ g) {
  int b = blockIdx.x, t = threadIdx.x;
  __shared__ float pooled[2 * CC];
  __shared__ float h1[CC / 4];
  pooled[t] = poolA[b * CC + t] * (1.f / NN);
  pooled[CC + t] = poolB[b * CC + t] * (1.f / NN);
  __syncthreads();
  if (t < CC / 4) {
    float acc = bg1[t];
    for (int j = 0; j < 2 * CC; ++j) acc += wg1[t * 2 * CC + j] * pooled[j];
    h1[t] = fmaxf(acc, 0.f);
  }
  __syncthreads();
  float acc = bg2[t];
  for (int j = 0; j < CC / 4; ++j) acc += wg2[t * (CC / 4) + j] * h1[j];
  g[b * CC + t] = 1.f / (1.f + __expf(-acc));
}

// ---------------------------------------------------------------------------
// K2: Gram matrix G[b] = A_b @ B_b^T
//     grid (b=16, colsplit=4), 512 thr = 16 waves; block tile 256 x 64
// ---------------------------------------------------------------------------
__global__ void __launch_bounds__(512, 1)
k_gram(const float* __restrict__ A, const float* __restrict__ B,
       float* __restrict__ G) {
  int b = blockIdx.x, cs = blockIdx.y;
  int t = threadIdx.x;
  int lane = t & 31, w = t >> 5;
  __shared__ __align__(16) bf16 At[256 * PITCH];
  __shared__ __align__(16) bf16 Bt[64 * PITCH];
  v8f acc[4];
#pragma unroll
  for (int j = 0; j < 4; ++j) acc[j] = v8f_zero();
  const float* Ab = A + (size_t)b * CC * NN;
  const float* Bb = B + ((size_t)b * CC + 64 * cs) * NN;
  int arow = t >> 1, akoff = (t & 1) * 16;
  int brow = t >> 3, bkoff = (t & 7) * 4;
  bf16* da = At + arow * PITCH + akoff;
  bf16* db = Bt + brow * PITCH + bkoff;
  for (int k0 = 0; k0 < NN; k0 += 32) {
    const float4* pa = (const float4*)(Ab + (size_t)arow * NN + k0 + akoff);
#pragma unroll
    for (int i = 0; i < 4; ++i) {
      float4 x = pa[i];
      da[4 * i + 0] = (bf16)x.x; da[4 * i + 1] = (bf16)x.y;
      da[4 * i + 2] = (bf16)x.z; da[4 * i + 3] = (bf16)x.w;
    }
    {
      float4 y = *(const float4*)(Bb + (size_t)brow * NN + k0 + bkoff);
      db[0] = (bf16)y.x; db[1] = (bf16)y.y; db[2] = (bf16)y.z; db[3] = (bf16)y.w;
    }
    __syncthreads();
    bf16x16 af = ldsA40(At, 16 * w, lane);
#pragma unroll
    for (int j = 0; j < 4; ++j)
      acc[j] = wmma_bf16(af, ldsB40(Bt, 16 * j, lane), acc[j]);
    __syncthreads();
  }
  float* Gb = G + (size_t)b * CC * CC;
  int mb = 16 * w + ((lane >> 4) ? 8 : 0);
  int ci = lane & 15;
#pragma unroll
  for (int j = 0; j < 4; ++j)
#pragma unroll
    for (int r = 0; r < 8; ++r)
      Gb[(size_t)(mb + r) * CC + 64 * cs + 16 * j + ci] = acc[j][r];
}

// ---------------------------------------------------------------------------
// K3: per (head, dir, batch): T=Wq_h@Gmat ; logits=s*T@Wk_h^T ; softmax ;
//     GM = diag(g) * (attn @ Wv_h)    (256 thr = 8 waves)
// ---------------------------------------------------------------------------
__global__ void __launch_bounds__(256, 1)
k_attn(const float* __restrict__ G,
       const float* __restrict__ wqA, const float* __restrict__ wkA,
       const float* __restrict__ wvA, const float* __restrict__ wqB,
       const float* __restrict__ wkB, const float* __restrict__ wvB,
       const float* __restrict__ g, float* __restrict__ GM) {
  int h = blockIdx.x, dir = blockIdx.y, b = blockIdx.z;
  int t = threadIdx.x;
  int lane = t & 31, w = t >> 5;
  const float* wq = dir ? wqB : wqA;
  const float* wk = dir ? wkA : wkB;
  const float* wv = dir ? wvA : wvB;
  const float* Gb = G + (size_t)b * CC * CC;

  __shared__ __align__(16) char smem[63616];
  bf16* Tl = (bf16*)smem;                          // 64*264 bf16 = 33792 B
  bf16* Gt = (bf16*)(smem + 33792);                // 256*40 bf16 = 20480 B
  float* Lg = (float*)(smem + 33792);              // overlays Gt: 64*68 f32
  bf16* Attn = (bf16*)(smem + 33792 + 20608);      // 64*72 bf16 = 9216 B

  int s = w >> 1;             // row strip (0..3) of the 64-row output
  int jb = (w & 1) * 8;       // col-tile base (0 or 8)

  // ---- Phase 1: T = Wq_h @ Gmat (Gmat = G or G^T), bf16 into Tl ----
  v8f acc[8];
#pragma unroll
  for (int j = 0; j < 8; ++j) acc[j] = v8f_zero();
  for (int c0 = 0; c0 < CC; c0 += 32) {
    if (dir == 0) {           // Gt[n][k] = G[c0+k][n]
      int k = t >> 3, nb = (t & 7) * 32;
      const float* p = Gb + (size_t)(c0 + k) * CC + nb;
      bf16* d = Gt + nb * PITCH + k;
#pragma unroll
      for (int i = 0; i < 32; ++i) d[i * PITCH] = (bf16)p[i];
    } else {                  // Gt[n][k] = G[n][c0+k]
      const float* p = Gb + (size_t)t * CC + c0;
      bf16* d = Gt + t * PITCH;
#pragma unroll
      for (int i = 0; i < 32; ++i) d[i] = (bf16)p[i];
    }
    __syncthreads();
    bf16x16 af = gloA(wq, CH * h + 16 * s, CC, c0, lane);
#pragma unroll
    for (int jj = 0; jj < 8; ++jj)
      acc[jj] = wmma_bf16(af, ldsB40(Gt, 16 * (jb + jj), lane), acc[jj]);
    __syncthreads();
  }
  {
    int mb = 16 * s + ((lane >> 4) ? 8 : 0), ci = lane & 15;
#pragma unroll
    for (int jj = 0; jj < 8; ++jj)
#pragma unroll
      for (int r = 0; r < 8; ++r)
        Tl[(mb + r) * 264 + 16 * (jb + jj) + ci] = (bf16)acc[jj][r];
  }
  __syncthreads();

  // ---- Phase 2: logits = 0.125 * T @ Wk_h^T (64x64) ----
  {
    v8f lacc[2];
    lacc[0] = v8f_zero(); lacc[1] = v8f_zero();
    int tt0 = 2 * w;
    int m = tt0 >> 2;  // same row-tile for both of this wave's tiles
    for (int c0 = 0; c0 < CC; c0 += 32) {
      bf16x16 af = ldsA(Tl, 16 * m, 264, c0, lane);
#pragma unroll
      for (int u = 0; u < 2; ++u) {
        int nn = (tt0 + u) & 3;
        lacc[u] = wmma_bf16(af, gloBrows(wk, CH * h + 16 * nn, CC, c0, lane), lacc[u]);
      }
    }
    int mb = 16 * m + ((lane >> 4) ? 8 : 0), ci = lane & 15;
#pragma unroll
    for (int u = 0; u < 2; ++u) {
      int nn = (tt0 + u) & 3;
#pragma unroll
      for (int r = 0; r < 8; ++r)
        Lg[(mb + r) * 68 + 16 * nn + ci] = 0.125f * lacc[u][r];
    }
  }
  __syncthreads();

  // ---- Phase 3: row softmax (64 rows x 64) ----
  if (t < 64) {
    float mx = -1e30f;
    for (int d = 0; d < 64; ++d) mx = fmaxf(mx, Lg[t * 68 + d]);
    float sum = 0.f;
    for (int d = 0; d < 64; ++d) sum += __expf(Lg[t * 68 + d] - mx);
    float inv = 1.f / sum;
    for (int d = 0; d < 64; ++d)
      Attn[t * 72 + d] = (bf16)(__expf(Lg[t * 68 + d] - mx) * inv);
  }
  __syncthreads();

  // ---- Phase 4: GM_h = diag(g) * (attn @ Wv_h)  (64x256 fp32 to ws) ----
  {
    v8f macc[8];
#pragma unroll
    for (int j = 0; j < 8; ++j) macc[j] = v8f_zero();
    for (int kk0 = 0; kk0 < CH; kk0 += 32) {
      bf16x16 af = ldsA(Attn, 16 * s, 72, kk0, lane);
#pragma unroll
      for (int jj = 0; jj < 8; ++jj)
        macc[jj] = wmma_bf16(af, gloBcols(wv, CH * h + kk0, CC, 16 * (jb + jj), lane),
                             macc[jj]);
    }
    float* GMb = GM + (size_t)(dir * BB + b) * CC * CC;
    int mb = 16 * s + ((lane >> 4) ? 8 : 0), ci = lane & 15;
#pragma unroll
    for (int jj = 0; jj < 8; ++jj)
#pragma unroll
      for (int r = 0; r < 8; ++r) {
        int c = CH * h + mb + r;
        GMb[(size_t)c * CC + 16 * (jb + jj) + ci] = g[b * CC + c] * macc[jj][r];
      }
  }
}

// ---------------------------------------------------------------------------
// K4: P[dir,b] = Wproj @ GM[dir,b]  (bf16 out; 512 thr = 16 waves)
// ---------------------------------------------------------------------------
__global__ void __launch_bounds__(512, 1)
k_proj(const float* __restrict__ GM, const float* __restrict__ wprojA,
       const float* __restrict__ wprojB, bf16* __restrict__ P) {
  int dir = blockIdx.x, b = blockIdx.y;
  int t = threadIdx.x;
  int lane = t & 31, w = t >> 5;
  const float* wp = dir ? wprojB : wprojA;
  const float* GMb = GM + (size_t)(dir * BB + b) * CC * CC;
  __shared__ __align__(16) bf16 Mt[256 * PITCH];
  v8f acc[16];
#pragma unroll
  for (int j = 0; j < 16; ++j) acc[j] = v8f_zero();
  int k = t >> 4, nb = (t & 15) * 16;
  bf16* d0 = Mt + nb * PITCH + k;
  for (int c0 = 0; c0 < CC; c0 += 32) {
    const float* p = GMb + (size_t)(c0 + k) * CC + nb;
#pragma unroll
    for (int i = 0; i < 16; ++i) d0[i * PITCH] = (bf16)p[i];
    __syncthreads();
    bf16x16 af = gloA(wp, 16 * w, CC, c0, lane);
#pragma unroll
    for (int j = 0; j < 16; ++j)
      acc[j] = wmma_bf16(af, ldsB40(Mt, 16 * j, lane), acc[j]);
    __syncthreads();
  }
  bf16* Pb = P + (size_t)(dir * BB + b) * CC * CC;
  int mb = 16 * w + ((lane >> 4) ? 8 : 0), ci = lane & 15;
#pragma unroll
  for (int j = 0; j < 16; ++j)
#pragma unroll
    for (int r = 0; r < 8; ++r)
      Pb[(size_t)(mb + r) * CC + 16 * j + ci] = (bf16)acc[j][r];
}

// ---------------------------------------------------------------------------
// K5: out[dir,b] = base + P[dir,b] @ X_flat  (dominant GEMM, fused residual)
//     grid (nchunk=16, dir=2, b=16), 512 thr = 16 waves, 256x256 out tile
// ---------------------------------------------------------------------------
__global__ void __launch_bounds__(512, 1)
k_apply(const float* __restrict__ A, const float* __restrict__ B,
        const bf16* __restrict__ P, float* __restrict__ out) {
  int nc = blockIdx.x, dir = blockIdx.y, b = blockIdx.z;
  int t = threadIdx.x;
  int lane = t & 31, w = t >> 5;
  const float* X = dir ? A : B;
  const float* Base = dir ? B : A;
  const float* Xb = X + (size_t)b * CC * NN;
  const float* Bb = Base + (size_t)b * CC * NN;
  const bf16* Pb = P + (size_t)(dir * BB + b) * CC * CC;
  float* ob = out + (size_t)dir * BB * CC * NN + (size_t)b * CC * NN;
  int n0 = nc * 256;
  __shared__ __align__(16) bf16 Xt[256 * PITCH];
  v8f acc[16];
#pragma unroll
  for (int j = 0; j < 16; ++j) acc[j] = v8f_zero();
  int k = t >> 4, nb = (t & 15) * 16;
  bf16* d0 = Xt + nb * PITCH + k;
  for (int c0 = 0; c0 < CC; c0 += 32) {
    const float* p = Xb + (size_t)(c0 + k) * NN + n0 + nb;
    if (c0 + 32 < CC)
      __builtin_prefetch(Xb + (size_t)(c0 + 32 + k) * NN + n0 + nb, 0, 1);
#pragma unroll
    for (int i = 0; i < 16; ++i) d0[i * PITCH] = (bf16)p[i];
    __syncthreads();
    bf16x16 af = gbfA(Pb, 16 * w, CC, c0, lane);
#pragma unroll
    for (int j = 0; j < 16; ++j)
      acc[j] = wmma_bf16(af, ldsB40(Xt, 16 * j, lane), acc[j]);
    __syncthreads();
  }
  int mb = 16 * w + ((lane >> 4) ? 8 : 0), ci = lane & 15;
#pragma unroll
  for (int j = 0; j < 16; ++j)
#pragma unroll
    for (int r = 0; r < 8; ++r) {
      size_t idx = (size_t)(mb + r) * NN + n0 + 16 * j + ci;
      ob[idx] = Bb[idx] + acc[j][r];
    }
}

// ---------------------------------------------------------------------------
extern "C" void kernel_launch(void* const* d_in, const int* in_sizes, int n_in,
                              void* d_out, int out_size, void* d_ws, size_t ws_size,
                              hipStream_t stream) {
  const float* A      = (const float*)d_in[0];
  const float* B      = (const float*)d_in[1];
  const float* wqA    = (const float*)d_in[2];
  const float* wkA    = (const float*)d_in[3];
  const float* wvA    = (const float*)d_in[4];
  const float* wqB    = (const float*)d_in[5];
  const float* wkB    = (const float*)d_in[6];
  const float* wvB    = (const float*)d_in[7];
  const float* wg1    = (const float*)d_in[8];
  const float* bg1    = (const float*)d_in[9];
  const float* wg2    = (const float*)d_in[10];
  const float* bg2    = (const float*)d_in[11];
  const float* wprojA = (const float*)d_in[12];
  const float* wprojB = (const float*)d_in[13];
  float* out = (float*)d_out;

  // workspace carve (floats):
  //   g[16*256] | poolA[16*256] | poolB[16*256] | G[16*256*256]
  //   | GM[2*16*256*256] | P (bf16, 2*16*256*256)
  float* g_ws    = (float*)d_ws;
  float* poolA_ws = g_ws + (size_t)BB * CC;
  float* poolB_ws = poolA_ws + (size_t)BB * CC;
  float* G_ws    = poolB_ws + (size_t)BB * CC;
  float* GM_ws   = G_ws + (size_t)BB * CC * CC;
  bf16*  P_ws    = (bf16*)(GM_ws + (size_t)2 * BB * CC * CC);

  hipMemsetAsync(poolA_ws, 0, (size_t)2 * BB * CC * sizeof(float), stream);
  k_pool<<<dim3(BB, 8), CC, 0, stream>>>(A, B, poolA_ws, poolB_ws);
  k_gate<<<BB, CC, 0, stream>>>(poolA_ws, poolB_ws, wg1, bg1, wg2, bg2, g_ws);
  k_gram<<<dim3(BB, 4), 512, 0, stream>>>(A, B, G_ws);
  k_attn<<<dim3(NH, 2, BB), 256, 0, stream>>>(G_ws, wqA, wkA, wvA, wqB, wkB, wvB,
                                              g_ws, GM_ws);
  k_proj<<<dim3(2, BB), 512, 0, stream>>>(GM_ws, wprojA, wprojB, P_ws);
  k_apply<<<dim3(BB, 2, BB), 512, 0, stream>>>(A, B, P_ws, out);
}
